// SparseLinear_24781961297974
// MI455X (gfx1250) — matrix operand order repaired and checked by reading, not compiled, not run on hardware
//
#include <hip/hip_runtime.h>

// ---------------- problem constants (match reference) ----------------
#define B_ROWS        2048
#define DIM           512        // K dimension of GEMM
#define VOCAB         100000
#define TOPK          32
#define NEGV          (-1e30f)

// ---------------- tiling ----------------
#define ROWS_PER_WG   16         // one WMMA M-tile of rows per workgroup
#define COLS_PER_PASS 128        // 8 waves x 16 cols
#define CHUNK         4096       // V columns per workgroup (phase 1)
#define NCHUNK        25         // ceil(100000/4096)
#define SUBS          2          // sub-owners per row (each scans 64 cols/pass)
#define CAND_PER_ROW  (NCHUNK * SUBS * TOPK)   // 1600 candidates per row
#define CAND_PAD      2048
#define P2_THREADS    256

typedef __bf16 v16bf __attribute__((ext_vector_type(16)));
typedef float  v8f   __attribute__((ext_vector_type(8)));

// =====================================================================
// Phase 0: one-shot W fp32 -> bf16 hi + residual lo split (streamed once;
// removes 128x-redundant conversion VALU from the GEMM hot loop)
// =====================================================================
__global__ __launch_bounds__(256)
void convert_w(const float* __restrict__ w,
               unsigned short* __restrict__ whi,
               unsigned short* __restrict__ wlo,
               long long n)   // n % 4 == 0
{
    long long i = ((long long)blockIdx.x * 256 + threadIdx.x) * 4;
    if (i + 3 >= n) return;
    const float4 v = *(const float4*)(w + i);
    union { __bf16 h[4]; uint2 u; } H, L;
#pragma unroll
    for (int e = 0; e < 4; ++e) {
        const float f  = (&v.x)[e];
        const __bf16 h = (__bf16)f;
        H.h[e] = h;
        L.h[e] = (__bf16)(f - (float)h);
    }
    *(uint2*)(whi + i) = H.u;
    *(uint2*)(wlo + i) = L.u;
}

// =====================================================================
// Phase 1: WMMA bf16x3 GEMM tile + fused bias/mask epilogue +
//          per-chunk running top-K (pure-LDS scan, 2 owners/row)
//   grid = (NCHUNK, B_ROWS/16), block = 256 (8 waves)
//   PRECONV=true : B fragments loaded directly as bf16 (zero-VALU hot loop)
//   PRECONV=false: on-the-fly fp32->bf16x2 split (workspace-starved fallback)
// =====================================================================
template<bool PRECONV>
__global__ __launch_bounds__(256)
void gemm_topk_phase1(const float* __restrict__ x,
                      const float* __restrict__ w,
                      const unsigned short* __restrict__ whi_,
                      const unsigned short* __restrict__ wlo_,
                      const float* __restrict__ bias,
                      const unsigned char* __restrict__ mask,
                      float* __restrict__ candV,
                      int*   __restrict__ candI)
{
    __shared__ __bf16 xhi[ROWS_PER_WG * DIM];                 // 16 KB
    __shared__ __bf16 xlo[ROWS_PER_WG * DIM];                 // 16 KB
    __shared__ float  blockVals[ROWS_PER_WG * COLS_PER_PASS]; // 8 KB

    const __bf16* whi = (const __bf16*)whi_;
    const __bf16* wlo = (const __bf16*)wlo_;

    const int tid  = threadIdx.x;
    const int wave = tid >> 5;
    const int lane = tid & 31;
    const int n    = lane & 15;        // matrix column-within-tile / A row
    const int half = lane >> 4;        // 0: lanes 0-15, 1: lanes 16-31

    const int rowBase    = blockIdx.y * ROWS_PER_WG;
    const int chunkStart = blockIdx.x * CHUNK;

    // ---- stage x tile (16x512 fp32) into LDS as bf16 hi + residual lo ----
    for (int i = tid; i < ROWS_PER_WG * DIM; i += 256) {
        float v   = x[(size_t)(rowBase + (i >> 9)) * DIM + (i & 511)];
        __bf16 hi = (__bf16)v;
        xhi[i] = hi;
        xlo[i] = (__bf16)(v - (float)hi);
    }
    __syncthreads();

    // ---- per-row running top-K state: 2 owner threads per row ----
    const bool owner = ((tid & 7) == 0);
    const int  orow  = tid >> 4;          // row within tile (0..15)
    const int  osub  = (tid >> 3) & 1;    // which 64-col half of the pass
    float tv[TOPK];
    int   ti[TOPK];
    float curMin = NEGV;
    int   minPos = 0;
    for (int j = 0; j < TOPK; ++j) { tv[j] = NEGV; ti[j] = 0; }

    // A-fragment offsets (16-bit A 16x32), contiguous K to match B:
    //   lanes 0-15: elems 0..15 -> K = ks+0..15
    //   lanes16-31: elems 0..15 -> K = ks+16..31
    const int aoff = half * 16;
    // B-fragment (32x16): lane covers 16 contiguous K at ks + half*16
    const int boff = half * 16;

    for (int cb = chunkStart; cb < chunkStart + CHUNK; cb += COLS_PER_PASS) {
        const int  gcol  = cb + wave * 16 + n;                 // this lane's vocab col
        const bool colOK = (gcol < VOCAB);
        const int  wrow  = colOK ? gcol : (VOCAB - 1);         // clamp, keep EXEC full

        const float*  wptr = w   + (size_t)wrow * DIM;
        const __bf16* whp  = whi + (size_t)wrow * DIM;
        const __bf16* wlp  = wlo + (size_t)wrow * DIM;

        // prefetch next pass's W rows into the cache hierarchy
        const int pf = gcol + COLS_PER_PASS;
        if (pf < VOCAB) {
            if (PRECONV) {
                __builtin_prefetch(whi + (size_t)pf * DIM, 0, 1);
                __builtin_prefetch(wlo + (size_t)pf * DIM, 0, 1);
            } else {
                __builtin_prefetch(w + (size_t)pf * DIM, 0, 1);
            }
        }

        // issue mask-byte + bias loads NOW so they overlap the whole k-loop
        unsigned char m8[8];
        float bcol = 0.f;
        if (colOK) {
            bcol = bias[gcol];
#pragma unroll
            for (int r = 0; r < 8; ++r)
                m8[r] = mask[(size_t)(rowBase + r + half * 8) * VOCAB + gcol];
        } else {
#pragma unroll
            for (int r = 0; r < 8; ++r) m8[r] = 0;
        }

        v8f acc = {0.f,0.f,0.f,0.f,0.f,0.f,0.f,0.f};

        for (int ks = 0; ks < DIM; ks += 32) {
            // ---- A fragments (hi/lo) from LDS ----
            const __bf16* p0 = &xhi[n * DIM + ks + aoff];
            const __bf16* q0 = &xlo[n * DIM + ks + aoff];
            v16bf ahi, alo;
#pragma unroll
            for (int e = 0; e < 8; ++e) {
                ahi[e]     = p0[e];
                ahi[e + 8] = p0[e + 8];
                alo[e]     = q0[e];
                alo[e + 8] = q0[e + 8];
            }
            // ---- B fragments (hi/lo) ----
            v16bf bhi, blo;
            if (PRECONV) {
                // direct 32-byte bf16 loads: no conversion VALU in hot loop
                bhi = *(const v16bf*)(whp + ks + boff);
                blo = *(const v16bf*)(wlp + ks + boff);
            } else {
                const float* wp = wptr + ks + boff;   // 16 contiguous K values
#pragma unroll
                for (int e = 0; e < 16; ++e) {
                    float v  = wp[e];
                    __bf16 h = (__bf16)v;
                    bhi[e]   = h;
                    blo[e]   = (__bf16)(v - (float)h);
                }
            }
            // ---- bf16x3 split-product accumulation (fp32-class accuracy) ----
            acc = __builtin_amdgcn_wmma_f32_16x16x32_bf16(false, ahi, false, bhi,
                                                          (short)0, acc, false, false);
            acc = __builtin_amdgcn_wmma_f32_16x16x32_bf16(false, alo, false, bhi,
                                                          (short)0, acc, false, false);
            acc = __builtin_amdgcn_wmma_f32_16x16x32_bf16(false, ahi, false, blo,
                                                          (short)0, acc, false, false);
        }

        // ---- epilogue: fuse bias + mask, dump C tile to LDS ----
        //      VGPR r, lane -> (M = r + half*8, N = wave*16 + n)
        __syncthreads();   // owners finished reading previous blockVals
#pragma unroll
        for (int r = 0; r < 8; ++r) {
            float v = (m8[r] != 0) ? (acc[r] + bcol) : NEGV;
            blockVals[(r + half * 8) * COLS_PER_PASS + wave * 16 + n] = v;
        }
        __syncthreads();

        // ---- selection: pure-LDS scan, float4-vectorized, 64 cols/owner ----
        if (owner) {
            const float4* rp4 =
                (const float4*)&blockVals[orow * COLS_PER_PASS + osub * 64];
            const int gbase = cb + osub * 64;
#pragma unroll 4
            for (int q = 0; q < 16; ++q) {
                const float4 v4 = rp4[q];
#pragma unroll
                for (int e = 0; e < 4; ++e) {
                    const float v = (&v4.x)[e];
                    if (v > curMin) {                 // NEG entries never pass
                        tv[minPos] = v; ti[minPos] = gbase + q * 4 + e;
                        curMin = tv[0]; minPos = 0;
                        for (int j = 1; j < TOPK; ++j)
                            if (tv[j] < curMin) { curMin = tv[j]; minPos = j; }
                    }
                }
            }
        }
    }

    // ---- emit per-(row, chunk, sub) candidate list ----
    if (owner) {
        const size_t base =
            ((((size_t)(rowBase + orow)) * NCHUNK + blockIdx.x) * SUBS + osub) * TOPK;
        for (int j = 0; j < TOPK; ++j) { candV[base + j] = tv[j]; candI[base + j] = ti[j]; }
    }
}

// =====================================================================
// Phase 2: merge 1600 candidates/row -> final sorted top-32
//   grid = B_ROWS blocks of 256; iterative parallel arg-max with
//   lowest-index tie-break (matches jax.lax.top_k ordering)
// =====================================================================
__global__ __launch_bounds__(P2_THREADS)
void topk_merge_phase2(const float* __restrict__ candV,
                       const int*   __restrict__ candI,
                       float* __restrict__ outV,
                       float* __restrict__ outI)
{
    __shared__ float sv[CAND_PAD];
    __shared__ int   si[CAND_PAD];
    __shared__ float rv[P2_THREADS];
    __shared__ int   rp[P2_THREADS];

    const int b   = blockIdx.x;
    const int tid = threadIdx.x;
    const size_t base = (size_t)b * CAND_PER_ROW;

    for (int i = tid; i < CAND_PAD; i += P2_THREADS) {
        if (i < CAND_PER_ROW) { sv[i] = candV[base + i]; si[i] = candI[base + i]; }
        else                  { sv[i] = -3.4e38f;        si[i] = 0x7fffffff;      }
    }
    __syncthreads();

    for (int it = 0; it < TOPK; ++it) {
        float bestV = -3.4e38f;
        int   bestP = 0;
        for (int i = tid; i < CAND_PAD; i += P2_THREADS) {
            const float v = sv[i];
            if (v > bestV || (v == bestV && si[i] < si[bestP])) { bestV = v; bestP = i; }
        }
        rv[tid] = bestV; rp[tid] = bestP;
        __syncthreads();
        for (int s = P2_THREADS / 2; s > 0; s >>= 1) {
            if (tid < s) {
                const float v2 = rv[tid + s]; const int p2 = rp[tid + s];
                if (v2 > rv[tid] || (v2 == rv[tid] && si[p2] < si[rp[tid]])) {
                    rv[tid] = v2; rp[tid] = p2;
                }
            }
            __syncthreads();
        }
        if (tid == 0) {
            const int p = rp[0];
            outV[(size_t)b * TOPK + it] = rv[0];
            outI[(size_t)b * TOPK + it] = (float)si[p];  // index as numeric value
            sv[p] = -3.4e38f;                            // exclude from later rounds
        }
        __syncthreads();
    }
}

// =====================================================================
extern "C" void kernel_launch(void* const* d_in, const int* in_sizes, int n_in,
                              void* d_out, int out_size, void* d_ws, size_t ws_size,
                              hipStream_t stream)
{
    const float*         x    = (const float*)d_in[0];
    const float*         w    = (const float*)d_in[1];
    const float*         bias = (const float*)d_in[2];
    const unsigned char* mask = (const unsigned char*)d_in[3];
    // d_in[4] is k (== 32), compile-time constant here

    // workspace layout: [candV][candI][whi][wlo]
    const size_t candElems = (size_t)B_ROWS * CAND_PER_ROW;
    const size_t wElems    = (size_t)VOCAB * DIM;
    char* p = (char*)d_ws;
    float*          candV = (float*)p;                  p += candElems * sizeof(float);
    int*            candI = (int*)p;                    p += candElems * sizeof(int);
    unsigned short* whi   = (unsigned short*)p;         p += wElems * sizeof(unsigned short);
    unsigned short* wlo   = (unsigned short*)p;

    const size_t needPre = candElems * 8 + wElems * 4;  // ~226 MB
    const bool preconv = (ws_size >= needPre);

    dim3 g1(NCHUNK, B_ROWS / ROWS_PER_WG);   // 25 x 128 workgroups

    if (preconv) {
        const long long n = (long long)wElems;
        convert_w<<<(unsigned)(n / 4 / 256), 256, 0, stream>>>(w, whi, wlo, n);
        gemm_topk_phase1<true><<<g1, 256, 0, stream>>>(x, w, whi, wlo, bias, mask,
                                                       candV, candI);
    } else {
        gemm_topk_phase1<false><<<g1, 256, 0, stream>>>(x, w, whi, wlo, bias, mask,
                                                        candV, candI);
    }

    float* outV = (float*)d_out;
    float* outI = outV + (size_t)B_ROWS * TOPK;
    topk_merge_phase2<<<B_ROWS, P2_THREADS, 0, stream>>>(candV, candI, outV, outI);
}
